// SharedGroupGRU_19636590477501
// MI455X (gfx1250) — matrix operand order repaired and checked by reading, not compiled, not run on hardware
//
#include <hip/hip_runtime.h>
#include <math.h>

// ---------------- problem constants ----------------
#define NB      16384          // N * NUM_HIDDEN rows
#define HD      128            // IN == HID
#define RULES   16
#define G3      384            // 3 * HID
#define KEYD    64
#define X_ELE   (NB * HD)      // 2,097,152
#define W_ELE   (RULES * G3 * HD) // 786,432
#define K_ELE   (KEYD * HD)    // 8,192
#define STG_LD  136            // padded half stride for LDS staging

typedef _Float16 v16h __attribute__((ext_vector_type(16)));
typedef _Float16 h8   __attribute__((ext_vector_type(8)));
typedef _Float16 h2   __attribute__((ext_vector_type(2)));
typedef float    v8f  __attribute__((ext_vector_type(8)));

__device__ __forceinline__ v8f wmma_f16(v16h a, v16h b, v8f c) {
  // D = A(16x32 f16) * B(32x16 f16) + C(16x16 f32)
  return __builtin_amdgcn_wmma_f32_16x16x32_f16(false, a, false, b, (short)0, c,
                                                false, false);
}

// A operand (16x32, 16-bit): lane r(0..15): halves[0..7]=K(k0..k0+7), [8..15]=K(k0+16..k0+23)
// lane r+16: K(k0+8..k0+15) and K(k0+24..k0+31).  base points at row 0 of the M-tile.
__device__ __forceinline__ v16h loadA(const _Float16* base, int ldm, int k0) {
  const int lane = threadIdx.x & 31;
  const int r = lane & 15, hi = lane >> 4;
  const _Float16* p = base + (size_t)r * ldm + k0 + hi * 8;
  h8 lo = *(const h8*)p;
  h8 hh = *(const h8*)(p + 16);
  return __builtin_shufflevector(lo, hh, 0,1,2,3,4,5,6,7,8,9,10,11,12,13,14,15);
}

// B operand (32x16, 16-bit): lane n: halves[0..7]=K(k0+hi*8..+7), [8..15]=K(k0+16+hi*8..+7).
// Weight matrices are stored [N][K] row-major, so column n of B == row (n0+n) of W.
__device__ __forceinline__ v16h loadB(const _Float16* base, int ldm, int n0, int k0) {
  const int lane = threadIdx.x & 31;
  const int n = lane & 15, hi = lane >> 4;
  const _Float16* p = base + (size_t)(n0 + n) * ldm + k0 + hi * 8;
  h8 lo = *(const h8*)p;
  h8 hh = *(const h8*)(p + 16);
  return __builtin_shufflevector(lo, hh, 0,1,2,3,4,5,6,7,8,9,10,11,12,13,14,15);
}

__device__ __forceinline__ float sigm(float x) { return 1.0f / (1.0f + expf(-x)); }

// ---------------- K1: f32 -> f16 conversions (weights stay L2-resident) ----------------
__global__ void convert_kernel(const float* __restrict__ x, const float* __restrict__ h,
                               const float* __restrict__ Wih, const float* __restrict__ Whh,
                               const float* __restrict__ wread, const float* __restrict__ Wwr,
                               _Float16* __restrict__ xf, _Float16* __restrict__ hf,
                               _Float16* __restrict__ wihf, _Float16* __restrict__ whhf,
                               _Float16* __restrict__ wwrf, _Float16* __restrict__ wrtf) {
  int i = blockIdx.x * blockDim.x + threadIdx.x;
  if (i < X_ELE) { xf[i] = (_Float16)x[i]; hf[i] = (_Float16)h[i]; }
  if (i < W_ELE) { wihf[i] = (_Float16)Wih[i]; whhf[i] = (_Float16)Whh[i]; }
  if (i < K_ELE) {
    wwrf[i] = (_Float16)Wwr[i];                 // W_write already [KEY][HID]
    int k = i >> 7, hh = i & 127;               // transpose w_read [HID][KEY] -> [KEY][HID]
    wrtf[i] = (_Float16)wread[hh * KEYD + k];
  }
}

// ---------------- K2: fused grouped-GRU + attention-select ----------------
__global__ __launch_bounds__(256) void gru_fused_kernel(
    const float* __restrict__ hglob, const float* __restrict__ bih,
    const float* __restrict__ bhh, const float* __restrict__ gumbel,
    const _Float16* __restrict__ xf, const _Float16* __restrict__ hf,
    const _Float16* __restrict__ wihf, const _Float16* __restrict__ whhf,
    const _Float16* __restrict__ wwrf, const _Float16* __restrict__ wrtf,
    float* __restrict__ outH, float* __restrict__ outA) {
  __shared__ __align__(16) _Float16 stage[8][16 * STG_LD]; // per-wave hnext staging (A-layout src)
  __shared__ float hhf32[16][HD];                          // f32 h_prev tile
  __shared__ float hread[16][68];                          // h_read tile (padded)
  __shared__ float logitsS[16][RULES];
  __shared__ float attS[16][RULES];

  const int tid = threadIdx.x;
  const int wave = tid >> 5, lane = tid & 31;
  const int col = lane & 15, hi = lane >> 4;
  const int m0 = blockIdx.x * 16;

  // f32 h_prev tile into LDS (coalesced)
  for (int i = tid; i < 16 * HD; i += 256)
    hhf32[i >> 7][i & 127] = hglob[(size_t)m0 * HD + i];

  // Resident A fragments for x and h (K = 128 -> 4 chunks of 32)
  v16h xfrag[4], hfrag[4];
#pragma unroll
  for (int kc = 0; kc < 4; ++kc) {
    xfrag[kc] = loadA(xf + (size_t)m0 * HD, HD, kc * 32);
    hfrag[kc] = loadA(hf + (size_t)m0 * HD, HD, kc * 32);
  }

  const v8f vzero = {0.f, 0.f, 0.f, 0.f, 0.f, 0.f, 0.f, 0.f};

  // h_read = hh @ w_read  (one wave, 16 WMMAs; issue all 4 B frags up front)
  if (wave == 0) {
#pragma unroll
    for (int t = 0; t < 4; ++t) {
      v16h b0 = loadB(wrtf, HD, t * 16, 0);
      v16h b1 = loadB(wrtf, HD, t * 16, 32);
      v16h b2 = loadB(wrtf, HD, t * 16, 64);
      v16h b3 = loadB(wrtf, HD, t * 16, 96);
      v8f acc = vzero;
      acc = wmma_f16(hfrag[0], b0, acc);
      acc = wmma_f16(hfrag[1], b1, acc);
      acc = wmma_f16(hfrag[2], b2, acc);
      acc = wmma_f16(hfrag[3], b3, acc);
#pragma unroll
      for (int v = 0; v < 8; ++v) hread[v + 8 * hi][t * 16 + col] = acc[v];
    }
  }
  __syncthreads();

  h2 hnreg[2][8][4]; // packed f16 hnext, 2 rules x 8 h-tiles x 8 rows (pairs)

  for (int rr = 0; rr < 2; ++rr) {
    const int rule = wave * 2 + rr;
    const _Float16* wih_r = wihf + (size_t)rule * G3 * HD;
    const _Float16* whh_r = whhf + (size_t)rule * G3 * HD;

    for (int ht = 0; ht < 8; ++ht) {
      const int c0 = ht * 16;
      v8f aIR = vzero, aIZ = vzero, aIN = vzero;
      v8f aHR = vzero, aHZ = vzero, aHN = vzero;
#pragma unroll
      for (int kc = 0; kc < 4; ++kc) {
        const int k0 = kc * 32;
        // Materialize all 6 B fragments (12 x b128 loads) before the 6 WMMAs
        // so the scheduler can keep a deep load pipe instead of 1-deep chains.
        v16h bIR = loadB(wih_r, HD, c0,       k0);
        v16h bIZ = loadB(wih_r, HD, c0 + 128, k0);
        v16h bIN = loadB(wih_r, HD, c0 + 256, k0);
        v16h bHR = loadB(whh_r, HD, c0,       k0);
        v16h bHZ = loadB(whh_r, HD, c0 + 128, k0);
        v16h bHN = loadB(whh_r, HD, c0 + 256, k0);
        aIR = wmma_f16(xfrag[kc], bIR, aIR);
        aIZ = wmma_f16(xfrag[kc], bIZ, aIZ);
        aIN = wmma_f16(xfrag[kc], bIN, aIN);
        aHR = wmma_f16(hfrag[kc], bHR, aHR);
        aHZ = wmma_f16(hfrag[kc], bHZ, aHZ);
        aHN = wmma_f16(hfrag[kc], bHN, aHN);
      }
      const int g = rule * G3 + c0 + col;
      const float bIRs = bih[g], bIZs = bih[g + 128], bINs = bih[g + 256];
      const float bHRs = bhh[g], bHZs = bhh[g + 128], bHNs = bhh[g + 256];
#pragma unroll
      for (int v = 0; v < 8; ++v) {
        const int m = v + 8 * hi;
        float rg = sigm(aIR[v] + bIRs + aHR[v] + bHRs);
        float zg = sigm(aIZ[v] + bIZs + aHZ[v] + bHZs);
        float ng = tanhf(aIN[v] + bINs + rg * (aHN[v] + bHNs));
        float hn = (1.0f - zg) * ng + zg * hhf32[m][c0 + col];
        stage[wave][m * STG_LD + c0 + col] = (_Float16)hn;
        hnreg[rr][ht][v >> 1][v & 1] = (_Float16)hn;
      }
    }

    // make per-wave LDS staging visible to the WMMA A-operand reads
    asm volatile("s_wait_dscnt 0" ::: "memory");

    // write_key = hnext @ W_write^T  and  logit = <h_read, write_key>
    float lg[8];
#pragma unroll
    for (int v = 0; v < 8; ++v) lg[v] = 0.f;
#pragma unroll
    for (int t = 0; t < 4; ++t) {
      // group the 4 A (LDS) and 4 B (global) fragments ahead of the WMMAs
      v16h a0 = loadA(&stage[wave][0], STG_LD, 0);
      v16h a1 = loadA(&stage[wave][0], STG_LD, 32);
      v16h a2 = loadA(&stage[wave][0], STG_LD, 64);
      v16h a3 = loadA(&stage[wave][0], STG_LD, 96);
      v16h b0 = loadB(wwrf, HD, t * 16, 0);
      v16h b1 = loadB(wwrf, HD, t * 16, 32);
      v16h b2 = loadB(wwrf, HD, t * 16, 64);
      v16h b3 = loadB(wwrf, HD, t * 16, 96);
      v8f wk = vzero;
      wk = wmma_f16(a0, b0, wk);
      wk = wmma_f16(a1, b1, wk);
      wk = wmma_f16(a2, b2, wk);
      wk = wmma_f16(a3, b3, wk);
#pragma unroll
      for (int v = 0; v < 8; ++v)
        lg[v] += wk[v] * hread[v + 8 * hi][t * 16 + col];
    }
    // reduce over the 16 key-columns held across each lane-half
#pragma unroll
    for (int v = 0; v < 8; ++v) {
      float s = lg[v];
      s += __shfl_xor(s, 1, 32);
      s += __shfl_xor(s, 2, 32);
      s += __shfl_xor(s, 4, 32);
      s += __shfl_xor(s, 8, 32);
      lg[v] = s;
    }
    if (col == 0) {
#pragma unroll
      for (int v = 0; v < 8; ++v) logitsS[v + 8 * hi][rule] = lg[v];
    }
  }
  __syncthreads();

  // gumbel-softmax (tau=0.5, hard=True, straight-through) — lanes 0..15 of wave 0
  if (tid < 16) {
    const int m = tid;
    float sc[RULES];
    float mx = -3.4e38f;
    int am = 0;
    for (int r = 0; r < RULES; ++r) {
      sc[r] = (logitsS[m][r] + gumbel[(size_t)(m0 + m) * RULES + r]) * 2.0f;
      if (sc[r] > mx) { mx = sc[r]; am = r; } // strict > == first-max (jnp.argmax)
    }
    float se = 0.f;
    for (int r = 0; r < RULES; ++r) { sc[r] = expf(sc[r] - mx); se += sc[r]; }
    const float inv = 1.0f / se;
    for (int r = 0; r < RULES; ++r) {
      float s = sc[r] * inv;
      float a = (((r == am) ? 1.0f : 0.0f) + s) - s; // exact ST arithmetic
      attS[m][r] = a;
      outA[(size_t)(m0 + m) * RULES + r] = a;
    }
  }
  __syncthreads();

  // h_out: only the argmax rule has att != 0, so exactly one wave writes each row
  for (int rr = 0; rr < 2; ++rr) {
    const int rule = wave * 2 + rr;
    for (int ht = 0; ht < 8; ++ht) {
#pragma unroll
      for (int v = 0; v < 8; ++v) {
        const int m = v + 8 * hi;
        const float coef = attS[m][rule];
        if (coef != 0.0f)
          outH[(size_t)(m0 + m) * HD + ht * 16 + col] =
              coef * (float)hnreg[rr][ht][v >> 1][v & 1];
      }
    }
  }
}

// ---------------- launcher ----------------
extern "C" void kernel_launch(void* const* d_in, const int* in_sizes, int n_in,
                              void* d_out, int out_size, void* d_ws, size_t ws_size,
                              hipStream_t stream) {
  const float* inp    = (const float*)d_in[0];
  const float* h      = (const float*)d_in[1];
  const float* Wih    = (const float*)d_in[2];
  const float* Whh    = (const float*)d_in[3];
  const float* bih    = (const float*)d_in[4];
  const float* bhh    = (const float*)d_in[5];
  const float* wread  = (const float*)d_in[6];
  const float* Wwrite = (const float*)d_in[7];
  const float* gum    = (const float*)d_in[8];

  _Float16* ws   = (_Float16*)d_ws;
  _Float16* xf   = ws;                 // 2,097,152 halves
  _Float16* hf   = xf + X_ELE;         // 2,097,152
  _Float16* wihf = hf + X_ELE;         // 786,432
  _Float16* whhf = wihf + W_ELE;       // 786,432
  _Float16* wwrf = whhf + W_ELE;       // 8,192
  _Float16* wrtf = wwrf + K_ELE;       // 8,192   (~11.5 MB total)

  float* outH = (float*)d_out;
  float* outA = outH + (size_t)NB * HD;

  convert_kernel<<<X_ELE / 256, 256, 0, stream>>>(inp, h, Wih, Whh, wread, Wwrite,
                                                  xf, hf, wihf, whhf, wwrf, wrtf);
  gru_fused_kernel<<<NB / 16, 256, 0, stream>>>(h, bih, bhh, gum, xf, hf, wihf,
                                                whhf, wwrf, wrtf, outH, outA);
}
